// Predictor_57561151701228
// MI455X (gfx1250) — matrix operand order repaired and checked by reading, not compiled, not run on hardware
//
#include <hip/hip_runtime.h>
#include <math.h>

#define SEQL 2111   // LOOKBACK + N_SAMPLES - 1

typedef __attribute__((ext_vector_type(16))) __bf16 v16bf;
typedef __attribute__((ext_vector_type(8)))  float  v8f;

// ---------------- bf16 helpers (RNE) ----------------
static __device__ inline __bf16 f2bf(float f) {
  union { float f; unsigned u; } in; in.f = f;
  unsigned u = in.u;
  unsigned short h;
  if ((u & 0x7F800000u) == 0x7F800000u) {
    h = (unsigned short)((u >> 16) | ((u & 0xFFFFu) ? 0x40u : 0u));
  } else {
    h = (unsigned short)((u + 0x7FFFu + ((u >> 16) & 1u)) >> 16);
  }
  union { unsigned short s; __bf16 b; } out; out.s = h;
  return out.b;
}
static __device__ inline float bf2f(__bf16 b) {
  union { __bf16 b; unsigned short s; } in; in.b = b;
  union { unsigned u; float f; } out; out.u = ((unsigned)in.s) << 16;
  return out.f;
}
static __device__ inline v8f wmma_bf16(v16bf a, v16bf b, v8f c) {
  return __builtin_amdgcn_wmma_f32_16x16x32_bf16(false, a, false, b, (short)0, c, false, false);
}

// ---------------- pack B (KxN fp32 row-major) into per-tile WMMA fragment order ----
// tile (kt,nt): 512 halves; element (lane,j): k = kt*32 + (j&8)*2 + (lane>>4)*8 + (j&7),
//                                             n = nt*16 + (lane&15). Zero pad k>=K.
__global__ void k_pack_b(const float* __restrict__ B, int K, int N,
                         __bf16* __restrict__ P, int Kp) {
  int NT = N >> 4;
  size_t total = (size_t)(Kp >> 5) * NT * 512;
  for (size_t idx = (size_t)blockIdx.x * blockDim.x + threadIdx.x; idx < total;
       idx += (size_t)gridDim.x * blockDim.x) {
    size_t tile = idx >> 9;
    int within = (int)(idx & 511);
    int lane2 = within >> 4;
    int j = within & 15;
    int kt = (int)(tile / NT);
    int nt = (int)(tile % NT);
    int k = kt * 32 + (j & 8) * 2 + (lane2 >> 4) * 8 + (j & 7);
    int n = nt * 16 + (lane2 & 15);
    float v = (k < K) ? B[(size_t)k * N + n] : 0.f;
    P[idx] = f2bf(v);
  }
}

__global__ void k_f2bf(const float* __restrict__ src, __bf16* __restrict__ dst, size_t n) {
  for (size_t i = (size_t)blockIdx.x * blockDim.x + threadIdx.x; i < n;
       i += (size_t)gridDim.x * blockDim.x)
    dst[i] = f2bf(src[i]);
}

__global__ void k_zero_h(float* __restrict__ h, __bf16* __restrict__ hb) {
  int i = blockIdx.x * blockDim.x + threadIdx.x;
  if (i < 64 * 512) { h[i] = 0.f; hb[i] = f2bf(0.f); }
}

// ---------------- A-matrix prep (dequant + hf gather), bf16, K padded ----------------
__global__ void k_prep_top(const int* __restrict__ seq, const float* __restrict__ hf,
                           __bf16* __restrict__ Ap /*2048x64*/, __bf16* __restrict__ Ah /*2048x32*/) {
  int i = blockIdx.x * blockDim.x + threadIdx.x;
  if (i < 2048 * 64) {
    int r = i >> 6, c = i & 63;
    int b = r >> 5, t = r & 31;
    float s = (float)seq[b * SEQL + t * 64 + c];
    Ap[i] = f2bf(2.f * (s * (1.f / 128.f) - 1.f));
  }
  if (i < 2048 * 32) {
    int r = i >> 5, c = i & 31;
    Ah[i] = f2bf(c < 16 ? hf[r * 16 + c] : 0.f);
  }
}

__global__ void k_prep_bot(const int* __restrict__ seq, const float* __restrict__ hf,
                           __bf16* __restrict__ Ap /*8192x32*/, __bf16* __restrict__ Ah /*8192x32*/) {
  int i = blockIdx.x * blockDim.x + threadIdx.x;
  if (i >= 8192 * 32) return;
  int r = i >> 5, c = i & 31;
  int b = r >> 7, tb = r & 127;
  float pv = 0.f, hv = 0.f;
  if (c < 16) {
    float s = (float)seq[b * SEQL + 48 + tb * 16 + c];
    pv = 2.f * (s * (1.f / 128.f) - 1.f);
    hv = hf[(b * 32 + (tb >> 2)) * 16 + c];   // repeat(hf, RATIO=4)
  }
  Ap[i] = f2bf(pv);
  Ah[i] = f2bf(hv);
}

// ---------------- hf features: Burg LPC + LSF via Durand-Kerner + residual RMS ------
static __device__ void dk_roots(const float* c, int n, float* ang, int* na) {
  float mc[17];
  float inv = 1.f / c[0];
  for (int i = 0; i <= n; ++i) mc[i] = c[i] * inv;
  float zr[16], zi[16];
  float pr = 1.f, pi = 0.f;
  for (int i = 0; i < n; ++i) {          // z_i = (0.4+0.9i)^(i+1)
    float nr = pr * 0.4f - pi * 0.9f;
    float ni = pr * 0.9f + pi * 0.4f;
    pr = nr; pi = ni;
    zr[i] = pr; zi[i] = pi;
  }
  for (int iter = 0; iter < 64; ++iter) {
    for (int i = 0; i < n; ++i) {
      float vr = 1.f, vi = 0.f;          // monic Horner
      for (int k2 = 1; k2 <= n; ++k2) {
        float tr = vr * zr[i] - vi * zi[i] + mc[k2];
        vi = vr * zi[i] + vi * zr[i];
        vr = tr;
      }
      float dr = 1.f, di = 0.f;
      for (int j = 0; j < n; ++j) {
        if (j == i) continue;
        float er = zr[i] - zr[j], ei = zi[i] - zi[j];
        float tr = dr * er - di * ei;
        di = dr * ei + di * er;
        dr = tr;
      }
      float d2 = dr * dr + di * di + 1e-30f;
      float qr = (vr * dr + vi * di) / d2;
      float qi = (vi * dr - vr * di) / d2;
      zr[i] -= qr; zi[i] -= qi;
    }
  }
  for (int i = 0; i < n; ++i) ang[(*na)++] = fabsf(atan2f(zi[i], zr[i]));
}

__global__ void k_hf(const int* __restrict__ seq, const float* __restrict__ noise,
                     float* __restrict__ hf) {
  int row = blockIdx.x * blockDim.x + threadIdx.x;
  if (row >= 2048) return;
  int b = row >> 5, t = row & 31;
  const float TWO_PI = 6.28318530717958647692f;
  const float NSTD = 0.0031622776601683794f;    // sqrt(1e-5)
  float x[64];
  for (int n = 0; n < 64; ++n) {
    float s = (float)seq[b * SEQL + t * 64 + n];
    float d = 2.f * (s * (1.f / 128.f) - 1.f);
    float wn = 0.5f - 0.5f * cosf(TWO_PI * (float)n / 63.f);
    x[n] = (d + NSTD * noise[(size_t)row * 64 + n]) * wn;
  }
  // Burg, ORDER=15, N=64
  float fwd[63], bwd[63], a[16];
  float den = 1e-12f;
  for (int i = 0; i < 63; ++i) {
    fwd[i] = x[i + 1]; bwd[i] = x[i];
    den += fwd[i] * fwd[i] + bwd[i] * bwd[i];
  }
  a[0] = 1.f;
  for (int i = 1; i < 16; ++i) a[i] = 0.f;
  for (int it = 0; it < 15; ++it) {
    float num = 0.f;
    for (int j = 0; j < 63; ++j) num += bwd[j] * fwd[j];
    float k = -2.f * num / den;
    float an[16];
    for (int m = 0; m < 16; ++m) {
      int idx = 15 - ((m + 14 - it) & 15);    // == a + k*roll(a[::-1], it+1-ORDER)
      an[m] = a[m] + k * a[idx];
    }
    for (int m = 0; m < 16; ++m) a[m] = an[m];
    int last = 62 - it;
    for (int j = 0; j < 63; ++j) {
      float tf = fwd[j];
      fwd[j] = tf + k * bwd[j];
      bwd[j] = bwd[j] + k * tf;
    }
    den = (1.f - k * k) * den - fwd[0] * fwd[0] - bwd[last] * bwd[last];
    for (int j = 0; j < 62; ++j) fwd[j] = (j < last) ? fwd[j + 1] : 0.f;
    fwd[62] = 0.f;
    for (int j = last; j < 63; ++j) bwd[j] = 0.f;
  }
  // LSF polynomials
  float a1[17];
  for (int i = 0; i < 16; ++i) a1[i] = a[i];
  a1[16] = 0.f;
  float P1[17], Q1[17];
  for (int k = 0; k < 17; ++k) { P1[k] = a1[k] - a1[16 - k]; Q1[k] = a1[k] + a1[16 - k]; }
  float P[15];
  P[0] = P1[0]; P[1] = P1[1];
  for (int k = 2; k < 15; ++k) P[k] = P1[k] + P[k - 2];
  float ang[30]; int na = 0;
  dk_roots(P, 14, ang, &na);
  dk_roots(Q1, 16, ang, &na);
  for (int i = 1; i < 30; ++i) {               // insertion sort ascending
    float v = ang[i]; int j = i - 1;
    while (j >= 0 && ang[j] > v) { ang[j + 1] = ang[j]; --j; }
    ang[j + 1] = v;
  }
  // residual rms
  float s2 = 0.f;
  for (int l = 0; l < 64; ++l) {
    float r = 0.f;
    int kmax = l < 15 ? l : 15;
    for (int k = 0; k <= kmax; ++k) r += a[k] * x[l - k];
    s2 += r * r;
  }
  float rms = sqrtf(s2 / 64.f);
  for (int q = 0; q < 15; ++q) hf[row * 16 + q] = ang[2 * q];
  hf[row * 16 + 15] = rms;
}

// ---------------- generic WMMA GEMM: one wave per 16x64 tile ------------------------
// C[M,N] = act( A[M,Kp]bf16 @ Bp(packed) + bias + Cadd + Cadd2 ); optional bf16 out.
__global__ void k_gemm(const __bf16* __restrict__ A, int lda,
                       const __bf16* __restrict__ Bp,
                       const float* __restrict__ bias,
                       const float* __restrict__ Cadd,
                       const float* __restrict__ Cadd2,
                       float* __restrict__ C,
                       __bf16* __restrict__ Cbf,
                       int M, int N, int Kp, int act) {
  __shared__ __align__(16) __bf16 sA[16 * 32];
  int lane = threadIdx.x;
  int m0 = blockIdx.y * 16;
  int ng = blockIdx.x;
  int n0 = ng * 64;
  int NT = N >> 4;
  v8f acc[4] = {};
  int row = m0 + (lane & 15);
  int half = lane >> 4;
  int nk = Kp >> 5;
  for (int kt = 0; kt < nk; ++kt) {
    const uint4* s4 = (const uint4*)(A + (size_t)row * lda + kt * 32 + half * 16);
    uint4* d4 = (uint4*)&sA[(lane & 15) * 32 + half * 16];
    d4[0] = s4[0]; d4[1] = s4[1];
    __syncthreads();
    union { v16bf v; uint4 q[2]; } af;
    int m = lane & 15;
    int b8 = (lane >> 4) * 8;
    af.q[0] = *(const uint4*)&sA[m * 32 + b8];        // k = b8..b8+7
    af.q[1] = *(const uint4*)&sA[m * 32 + 16 + b8];   // k = 16+b8..
#pragma unroll
    for (int i = 0; i < 4; ++i) {
      union { v16bf v; uint4 q[2]; } bfrag;
      const uint4* bp = (const uint4*)&Bp[((size_t)kt * NT + (ng * 4 + i)) * 512 + lane * 16];
      bfrag.q[0] = bp[0]; bfrag.q[1] = bp[1];
      acc[i] = wmma_bf16(af.v, bfrag.v, acc[i]);
    }
    __syncthreads();
  }
  int rbase = (lane >> 4) * 8;
  int ncol = lane & 15;
#pragma unroll
  for (int i = 0; i < 4; ++i) {
    int n = n0 + i * 16 + ncol;
#pragma unroll
    for (int r = 0; r < 8; ++r) {
      int m = m0 + rbase + r;
      float v = acc[i][r];
      if (bias)  v += bias[n];
      size_t off = (size_t)m * N + n;
      if (Cadd)  v += Cadd[off];
      if (Cadd2) v += Cadd2[off];
      if (act == 1) v = v > 0.f ? v : 0.f;
      if (C)   C[off] = v;
      if (Cbf) Cbf[off] = f2bf(v);
    }
  }
}

// ---------------- GRU gate elementwise ----------------------------------------------
__global__ void k_gate(const float* __restrict__ gx, int T, int t,
                       const float* __restrict__ gh,
                       float* __restrict__ h, __bf16* __restrict__ hb,
                       __bf16* __restrict__ hall) {
  int i = blockIdx.x * blockDim.x + threadIdx.x;
  if (i >= 64 * 512) return;
  int b = i >> 9, n = i & 511;
  const float* gxr = gx + ((size_t)b * T + t) * 1536;
  const float* ghr = gh + (size_t)b * 1536;
  float r = 1.f / (1.f + expf(-(gxr[n] + ghr[n])));
  float z = 1.f / (1.f + expf(-(gxr[512 + n] + ghr[512 + n])));
  float nn = tanhf(gxr[1024 + n] + r * ghr[1024 + n]);
  float hv = (1.f - z) * nn + z * h[i];
  h[i] = hv;
  __bf16 qb = f2bf(hv);
  hb[i] = qb;
  hall[((size_t)b * T + t) * 512 + n] = qb;
}

// ---------------- fused conv + cond + MLP + output ----------------------------------
static __device__ inline void layer512(const __bf16* sin_, __bf16* sout,
                                       const __bf16* __restrict__ pW,
                                       const float* __restrict__ bias,
                                       int lane, int w) {
  v8f acc[4] = {};
  int m = lane & 15;
  int b8 = (lane >> 4) * 8;
  for (int kt = 0; kt < 16; ++kt) {
    union { v16bf v; uint4 q[2]; } af;
    const __bf16* arow = &sin_[m * 512 + kt * 32 + b8];
    af.q[0] = *(const uint4*)arow;
    af.q[1] = *(const uint4*)(arow + 16);
#pragma unroll
    for (int i = 0; i < 4; ++i) {
      union { v16bf v; uint4 q[2]; } bfrag;
      const uint4* bp = (const uint4*)&pW[((size_t)kt * 32 + (w * 4 + i)) * 512 + lane * 16];
      bfrag.q[0] = bp[0]; bfrag.q[1] = bp[1];
      acc[i] = wmma_bf16(af.v, bfrag.v, acc[i]);
    }
  }
#pragma unroll
  for (int i = 0; i < 4; ++i) {
    int n = w * 64 + i * 16 + (lane & 15);
#pragma unroll
    for (int r = 0; r < 8; ++r) {
      int mr = (lane >> 4) * 8 + r;
      float v = acc[i][r] + bias[n];
      v = v > 0.f ? v : 0.f;
      sout[mr * 512 + n] = f2bf(v);
    }
  }
}

__global__ void __launch_bounds__(256)
k_final(const int* __restrict__ seq, const __bf16* __restrict__ emb,
        const __bf16* __restrict__ cond /*131072x512*/,
        const __bf16* __restrict__ pWc, const __bf16* __restrict__ pW1,
        const __bf16* __restrict__ pW2, const __bf16* __restrict__ pWo,
        const float* __restrict__ b_conv, const float* __restrict__ b1,
        const float* __restrict__ b2, const float* __restrict__ b_out,
        float* __restrict__ out) {
  __shared__ __align__(16) __bf16 se[31 * 256];
  __shared__ __align__(16) __bf16 sh1[16 * 512];
  __shared__ __align__(16) __bf16 sh2[16 * 512];
  int b = blockIdx.y;
  int t0 = blockIdx.x * 16;
  int tid = threadIdx.x, lane = tid & 31, w = tid >> 5;
  for (int idx = tid; idx < 31 * 256; idx += 256) {
    int s = idx >> 8, k = idx & 255;
    int qi = seq[b * SEQL + 48 + t0 + s];
    se[idx] = emb[qi * 256 + k];
  }
  __syncthreads();
  // conv: K = 16 taps x 256, N = 512, each wave owns 64 cols
  v8f acc[4] = {};
  int m = lane & 15;
  int b8 = (lane >> 4) * 8;
  for (int tap = 0; tap < 16; ++tap) {
    const __bf16* erow = &se[(m + tap) * 256];
    for (int ks = 0; ks < 8; ++ks) {
      union { v16bf v; uint4 q[2]; } af;
      af.q[0] = *(const uint4*)&erow[ks * 32 + b8];
      af.q[1] = *(const uint4*)&erow[ks * 32 + 16 + b8];
      int kt = tap * 8 + ks;
#pragma unroll
      for (int i = 0; i < 4; ++i) {
        union { v16bf v; uint4 q[2]; } bfrag;
        const uint4* bp = (const uint4*)&pWc[((size_t)kt * 32 + (w * 4 + i)) * 512 + lane * 16];
        bfrag.q[0] = bp[0]; bfrag.q[1] = bp[1];
        acc[i] = wmma_bf16(af.v, bfrag.v, acc[i]);
      }
    }
  }
#pragma unroll
  for (int i = 0; i < 4; ++i) {
    int n = w * 64 + i * 16 + (lane & 15);
#pragma unroll
    for (int r = 0; r < 8; ++r) {
      int mr = (lane >> 4) * 8 + r;
      size_t crow = ((size_t)b * 2048 + t0 + mr) * 512 + n;
      float v = acc[i][r] + b_conv[n] + bf2f(cond[crow]);
      v = v > 0.f ? v : 0.f;
      sh1[mr * 512 + n] = f2bf(v);
    }
  }
  __syncthreads();
  layer512(sh1, sh2, pW1, b1, lane, w);
  __syncthreads();
  layer512(sh2, sh1, pW2, b2, lane, w);
  __syncthreads();
  // output projection: N = 256, each wave owns 32 cols
  v8f oc[2] = {};
  for (int kt = 0; kt < 16; ++kt) {
    union { v16bf v; uint4 q[2]; } af;
    const __bf16* arow = &sh1[m * 512 + kt * 32 + b8];
    af.q[0] = *(const uint4*)arow;
    af.q[1] = *(const uint4*)(arow + 16);
#pragma unroll
    for (int i = 0; i < 2; ++i) {
      union { v16bf v; uint4 q[2]; } bfrag;
      const uint4* bp = (const uint4*)&pWo[((size_t)kt * 16 + (w * 2 + i)) * 512 + lane * 16];
      bfrag.q[0] = bp[0]; bfrag.q[1] = bp[1];
      oc[i] = wmma_bf16(af.v, bfrag.v, oc[i]);
    }
  }
#pragma unroll
  for (int i = 0; i < 2; ++i) {
    int n = w * 32 + i * 16 + (lane & 15);
#pragma unroll
    for (int r = 0; r < 8; ++r) {
      int mr = (lane >> 4) * 8 + r;
      out[((size_t)b * 2048 + t0 + mr) * 256 + n] = oc[i][r] + b_out[n];
    }
  }
}

// ---------------- host orchestration -------------------------------------------------
extern "C" void kernel_launch(void* const* d_in, const int* in_sizes, int n_in,
                              void* d_out, int out_size, void* d_ws, size_t ws_size,
                              hipStream_t stream) {
  (void)in_sizes; (void)n_in; (void)out_size; (void)ws_size;
  const int*   seq      = (const int*)  d_in[0];
  const float* noise    = (const float*)d_in[2];
  const float* W_in_top = (const float*)d_in[3];
  const float* W_hf_top = (const float*)d_in[4];
  const float* b_in_top = (const float*)d_in[5];
  const float* Wx_top   = (const float*)d_in[6];
  const float* bx_top   = (const float*)d_in[7];
  const float* Wh_top   = (const float*)d_in[8];
  const float* W_up_top = (const float*)d_in[9];
  const float* W_in_bot = (const float*)d_in[10];
  const float* W_hf_bot = (const float*)d_in[11];
  const float* b_in_bot = (const float*)d_in[12];
  const float* Wx_bot   = (const float*)d_in[13];
  const float* bx_bot   = (const float*)d_in[14];
  const float* Wh_bot   = (const float*)d_in[15];
  const float* W_up_bot = (const float*)d_in[16];
  const float* embed    = (const float*)d_in[17];
  const float* W_conv   = (const float*)d_in[18];
  const float* b_conv   = (const float*)d_in[19];
  const float* W1       = (const float*)d_in[20];
  const float* b1       = (const float*)d_in[21];
  const float* W2       = (const float*)d_in[22];
  const float* b2       = (const float*)d_in[23];
  const float* W_out    = (const float*)d_in[24];
  const float* b_out    = (const float*)d_in[25];
  float* out = (float*)d_out;

  char* p = (char*)d_ws;
  auto carve = [&](size_t bytes) -> char* {
    char* r = p; p += (bytes + 255) & ~(size_t)255; return r;
  };
  __bf16* pWinT = (__bf16*)carve((size_t)64 * 512 * 2);
  __bf16* pWhfT = (__bf16*)carve((size_t)32 * 512 * 2);
  __bf16* pWxT  = (__bf16*)carve((size_t)512 * 1536 * 2);
  __bf16* pWhT  = (__bf16*)carve((size_t)512 * 1536 * 2);
  __bf16* pWupT = (__bf16*)carve((size_t)512 * 2048 * 2);
  __bf16* pWinB = (__bf16*)carve((size_t)32 * 512 * 2);
  __bf16* pWhfB = (__bf16*)carve((size_t)32 * 512 * 2);
  __bf16* pWxB  = (__bf16*)carve((size_t)512 * 1536 * 2);
  __bf16* pWhB  = (__bf16*)carve((size_t)512 * 1536 * 2);
  __bf16* pWupB = (__bf16*)carve((size_t)512 * 8192 * 2);
  __bf16* pWc   = (__bf16*)carve((size_t)4096 * 512 * 2);
  __bf16* pW1   = (__bf16*)carve((size_t)512 * 512 * 2);
  __bf16* pW2   = (__bf16*)carve((size_t)512 * 512 * 2);
  __bf16* pWo   = (__bf16*)carve((size_t)512 * 256 * 2);
  __bf16* embB  = (__bf16*)carve((size_t)256 * 256 * 2);

  float*  hfb   = (float*) carve((size_t)2048 * 16 * 4);
  __bf16* AtP   = (__bf16*)carve((size_t)2048 * 64 * 2);
  __bf16* AtH   = (__bf16*)carve((size_t)2048 * 32 * 2);
  float*  inpT  = (float*) carve((size_t)2048 * 512 * 4);
  __bf16* inpTb = (__bf16*)carve((size_t)2048 * 512 * 2);
  float*  gxT   = (float*) carve((size_t)2048 * 1536 * 4);
  float*  hbuf  = (float*) carve((size_t)64 * 512 * 4);
  __bf16* hbufB = (__bf16*)carve((size_t)64 * 512 * 2);
  float*  gh    = (float*) carve((size_t)64 * 1536 * 4);
  __bf16* hTopB = (__bf16*)carve((size_t)2048 * 512 * 2);
  float*  condB = (float*) carve((size_t)2048 * 2048 * 4);
  __bf16* AbP   = (__bf16*)carve((size_t)8192 * 32 * 2);
  __bf16* AbH   = (__bf16*)carve((size_t)8192 * 32 * 2);
  float*  inpB  = (float*) carve((size_t)8192 * 512 * 4);
  __bf16* inpBb = (__bf16*)carve((size_t)8192 * 512 * 2);
  float*  gxB   = (float*) carve((size_t)8192 * 1536 * 4);
  __bf16* hBotB = (__bf16*)carve((size_t)8192 * 512 * 2);
  __bf16* condS = (__bf16*)carve((size_t)131072 * 512 * 2);

  auto pack = [&](const float* src, int K, int N, int Kp, __bf16* dst) {
    k_pack_b<<<dim3(256), 256, 0, stream>>>(src, K, N, dst, Kp);
  };
  pack(W_in_top, 64, 512, 64, pWinT);
  pack(W_hf_top, 16, 512, 32, pWhfT);
  pack(Wx_top, 512, 1536, 512, pWxT);
  pack(Wh_top, 512, 1536, 512, pWhT);
  pack(W_up_top, 512, 2048, 512, pWupT);
  pack(W_in_bot, 16, 512, 32, pWinB);
  pack(W_hf_bot, 16, 512, 32, pWhfB);
  pack(Wx_bot, 512, 1536, 512, pWxB);
  pack(Wh_bot, 512, 1536, 512, pWhB);
  pack(W_up_bot, 512, 8192, 512, pWupB);
  pack(W_conv, 4096, 512, 4096, pWc);
  pack(W1, 512, 512, 512, pW1);
  pack(W2, 512, 512, 512, pW2);
  pack(W_out, 512, 256, 512, pWo);
  k_f2bf<<<dim3(64), 256, 0, stream>>>(embed, embB, (size_t)256 * 256);

  k_hf<<<dim3(8), 256, 0, stream>>>(seq, noise, hfb);
  k_prep_top<<<dim3(512), 256, 0, stream>>>(seq, hfb, AtP, AtH);

  auto gemm = [&](const __bf16* A, int lda, const __bf16* Bp, const float* bias,
                  const float* Cadd, const float* Cadd2, float* C, __bf16* Cbf,
                  int M, int N, int Kp, int act) {
    k_gemm<<<dim3(N / 64, M / 16), 32, 0, stream>>>(A, lda, Bp, bias, Cadd, Cadd2,
                                                    C, Cbf, M, N, Kp, act);
  };

  // top tier
  gemm(AtP, 64, pWinT, b_in_top, nullptr, nullptr, inpT, nullptr, 2048, 512, 64, 0);
  gemm(AtH, 32, pWhfT, nullptr, inpT, nullptr, nullptr, inpTb, 2048, 512, 32, 0);
  gemm(inpTb, 512, pWxT, bx_top, nullptr, nullptr, gxT, nullptr, 2048, 1536, 512, 0);
  k_zero_h<<<dim3(128), 256, 0, stream>>>(hbuf, hbufB);
  for (int t = 0; t < 32; ++t) {
    gemm(hbufB, 512, pWhT, nullptr, nullptr, nullptr, gh, nullptr, 64, 1536, 512, 0);
    k_gate<<<dim3(128), 256, 0, stream>>>(gxT, 32, t, gh, hbuf, hbufB, hTopB);
  }
  gemm(hTopB, 512, pWupT, nullptr, nullptr, nullptr, condB, nullptr, 2048, 2048, 512, 0);

  // bottom tier (cond_bot reshape == plain row-major reinterpretation)
  k_prep_bot<<<dim3(1024), 256, 0, stream>>>(seq, hfb, AbP, AbH);
  gemm(AbP, 32, pWinB, b_in_bot, nullptr, nullptr, inpB, nullptr, 8192, 512, 32, 0);
  gemm(AbH, 32, pWhfB, nullptr, inpB, condB, nullptr, inpBb, 8192, 512, 32, 0);
  gemm(inpBb, 512, pWxB, bx_bot, nullptr, nullptr, gxB, nullptr, 8192, 1536, 512, 0);
  k_zero_h<<<dim3(128), 256, 0, stream>>>(hbuf, hbufB);
  for (int t = 0; t < 128; ++t) {
    gemm(hbufB, 512, pWhB, nullptr, nullptr, nullptr, gh, nullptr, 64, 1536, 512, 0);
    k_gate<<<dim3(128), 256, 0, stream>>>(gxB, 128, t, gh, hbuf, hbufB, hBotB);
  }
  gemm(hBotB, 512, pWupB, nullptr, nullptr, nullptr, nullptr, condS, 8192, 8192, 512, 0);

  // fused conv + cond + MLP + output
  k_final<<<dim3(128, 64), 256, 0, stream>>>(seq, embB, condS, pWc, pW1, pW2, pWo,
                                             b_conv, b1, b2, b_out, out);
}
